// Denoiser_80384607912004
// MI455X (gfx1250) — matrix operand order repaired and checked by reading, not compiled, not run on hardware
//
#include <hip/hip_runtime.h>
#include <stdint.h>

typedef __attribute__((ext_vector_type(16))) __bf16 v16bf;
typedef __attribute__((ext_vector_type(8)))  __bf16 v8bf;
typedef __attribute__((ext_vector_type(8)))  float  v8f;

#define WAVES 4
#define SI 128
#define SE 32
#define SO 64
#define ANGS 7

__device__ __forceinline__ float sigm(float x) { return 1.0f / (1.0f + __expf(-x)); }

// ---------------------------------------------------------------------------
// zero workspace accumulators
__global__ void zero_f(float* __restrict__ p, int n) {
    int id = blockIdx.x * blockDim.x + threadIdx.x;
    if (id < n) p[id] = 0.0f;
}

// ---------------------------------------------------------------------------
// Convert a [KREAL,64] f32 weight matrix into bf16 WMMA B fragments,
// zero-padded to KT*32 rows.  Fragment layout: frag[((kt*4+nt)*32+lane)*16+j]
// holds B[kt*32 + (j>>3)*16 + (lane>>4)*8 + (j&7)][nt*16 + (lane&15)]  (per ISA B layout).
__global__ void convert_w(const float* __restrict__ W, __bf16* __restrict__ out,
                          int KT, int KREAL) {
    int id = blockIdx.x * blockDim.x + threadIdx.x;
    int total = KT * 4 * 32 * 16;
    if (id >= total) return;
    int j    = id & 15;
    int lane = (id >> 4) & 31;
    int nt   = (id >> 9) & 3;
    int kt   = id >> 11;
    int k = kt * 32 + ((j >> 3) << 4) + ((lane >> 4) << 3) + (j & 7);
    int n = nt * 16 + (lane & 15);
    float v = (k < KREAL) ? W[k * 64 + n] : 0.0f;
    out[id] = (__bf16)v;
}

// ---------------------------------------------------------------------------
// Main per-edge kernel: one wave handles 16 edges through all 3 GVP layers.
__global__ __launch_bounds__(32 * WAVES, 1)
void gvp_edge_kernel(const float* __restrict__ node_s, const float* __restrict__ node_v,
                     const int* __restrict__ eidx, const float* __restrict__ edge_s,
                     const float* __restrict__ edge_v,
                     const float* __restrict__ w1_wh, const float* __restrict__ w1_wsb,
                     const float* __restrict__ w1_wv, const float* __restrict__ w1_wsv,
                     const float* __restrict__ w1_wsvb,
                     const float* __restrict__ w2_wh, const float* __restrict__ w2_wsb,
                     const float* __restrict__ w2_wv, const float* __restrict__ w2_wsv,
                     const float* __restrict__ w2_wsvb,
                     const float* __restrict__ w3_wh, const float* __restrict__ w3_wsb,
                     const float* __restrict__ w3_wv, const float* __restrict__ w3_wsv,
                     const float* __restrict__ w3_wsvb,
                     const __bf16* __restrict__ w1f, const __bf16* __restrict__ w2f,
                     const __bf16* __restrict__ w3f,
                     float* __restrict__ s_acc, float* __restrict__ v_acc,
                     float* __restrict__ cnt, int E_)
{
    // A-stage: bf16 rows of 384 (cols 0..351 used by WMMA, cols 128..255 reused
    // as 64 f32 "raw s" scratch between layers).
    __shared__ alignas(32) __bf16 Ast[WAVES][16][384];
    __shared__ int sidx[WAVES][16], didx[WAVES][16];

    const int w    = threadIdx.x >> 5;
    const int lane = threadIdx.x & 31;
    const int tile = blockIdx.x * WAVES + w;
    const int e0   = tile * 16;

    if (lane < 16) {
        sidx[w][lane] = eidx[e0 + lane];
        didx[w][lane] = eidx[E_ + e0 + lane];
    }
    __syncthreads();

    const int em = lane >> 1, p = lane & 1;   // 2 lanes per edge
    const int e  = e0 + em;
    const int sj = sidx[w][em], di = didx[w][em];

    // ---- stage scalar message [s_j(128) | edge_s(32) | s_i(128)] as bf16 ----
    for (int k = 0; k < 144; ++k) {
        int col = p * 144 + k;
        float v;
        if (col < SI)           v = node_s[sj * SI + col];
        else if (col < SI + SE) v = edge_s[e * SE + (col - SI)];
        else                    v = node_s[di * SI + (col - SI - SE)];
        Ast[w][em][col] = (__bf16)v;
    }
    if (p) {
        #pragma unroll
        for (int c = 321; c < 352; ++c) Ast[w][em][c] = (__bf16)0.0f;
    }

    // ---- vh = v_msg^T @ w1_wh for my h-range; vn -> A cols 288..320 ----
    const int nh = 17 - p, hb = p * 17;
    float vh[17][3];
    #pragma unroll
    for (int j = 0; j < 17; ++j) { vh[j][0] = 0.f; vh[j][1] = 0.f; vh[j][2] = 0.f; }
    for (int iv = 0; iv < 33; ++iv) {
        float vm0, vm1, vm2;
        if (iv < 16)       { const float* q = &node_v[(size_t)sj * 48 + iv * 3];        vm0=q[0]; vm1=q[1]; vm2=q[2]; }
        else if (iv == 16) { const float* q = &edge_v[(size_t)e * 3];                   vm0=q[0]; vm1=q[1]; vm2=q[2]; }
        else               { const float* q = &node_v[(size_t)di * 48 + (iv - 17) * 3]; vm0=q[0]; vm1=q[1]; vm2=q[2]; }
        #pragma unroll
        for (int j = 0; j < 17; ++j) {
            if (j < nh) {
                float ww = w1_wh[iv * 33 + hb + j];
                vh[j][0] = fmaf(vm0, ww, vh[j][0]);
                vh[j][1] = fmaf(vm1, ww, vh[j][1]);
                vh[j][2] = fmaf(vm2, ww, vh[j][2]);
            }
        }
    }
    float vp1[12];
    #pragma unroll
    for (int t = 0; t < 12; ++t) vp1[t] = 0.f;
    #pragma unroll
    for (int j = 0; j < 17; ++j) {
        if (j < nh) {
            float nn = vh[j][0]*vh[j][0] + vh[j][1]*vh[j][1] + vh[j][2]*vh[j][2];
            float vn = sqrtf(fmaxf(nn, 1e-8f));
            Ast[w][em][288 + hb + j] = (__bf16)vn;
            int h = hb + j;
            #pragma unroll
            for (int c = 0; c < 4; ++c) {
                float wc = w1_wv[h * 4 + c];
                vp1[c*3+0] = fmaf(vh[j][0], wc, vp1[c*3+0]);
                vp1[c*3+1] = fmaf(vh[j][1], wc, vp1[c*3+1]);
                vp1[c*3+2] = fmaf(vh[j][2], wc, vp1[c*3+2]);
            }
        }
    }
    #pragma unroll
    for (int t = 0; t < 12; ++t) vp1[t] += __shfl_xor(vp1[t], 1, 32);
    __syncthreads();

    // ---- WMMA helpers ----
    const int mr = lane & 15, hi = lane >> 4;
    const __bf16* arow = &Ast[w][mr][0];
    auto load_a = [&](int kt) -> v16bf {
        v8bf lo = *(const v8bf*)(arow + kt * 32 + hi * 8);       // K = kt*32+hi*8..+7
        v8bf hh = *(const v8bf*)(arow + kt * 32 + 16 + hi * 8);  // K = kt*32+16+hi*8..+7
        v16bf a;
        #pragma unroll
        for (int j = 0; j < 8; ++j) { a[j] = lo[j]; a[8 + j] = hh[j]; }
        return a;
    };

    // ================= GVP layer 1: [16,352]x[352,64] ======================
    v8f acc[4] = {};
    #pragma unroll
    for (int kt = 0; kt < 11; ++kt) {
        v16bf a = load_a(kt);
        #pragma unroll
        for (int nt = 0; nt < 4; ++nt) {
            v16bf b = *(const v16bf*)(w1f + (size_t)((kt * 4 + nt) * 32 + lane) * 16);
            acc[nt] = __builtin_amdgcn_wmma_f32_16x16x32_bf16(
                false, a, false, b, (short)0, acc[nt], false, false);
        }
    }
    #pragma unroll
    for (int nt = 0; nt < 4; ++nt)
        #pragma unroll
        for (int r = 0; r < 8; ++r) {
            int col = nt * 16 + mr, m = r + 8 * hi;
            float val = acc[nt][r] + w1_wsb[col];
            ((float*)&Ast[w][m][128])[col] = val;               // raw s1
            Ast[w][m][col] = (__bf16)fmaxf(val, 0.f);           // relu(s1) -> A2
        }
    __syncthreads();

    // ---- per-edge: gate1, v1, vh2, vn2, vector partials for layer 2 ----
    float vp2[12];
    {
        const float* s1 = (const float*)&Ast[w][em][128];
        float tg[4] = { w1_wsvb[0], w1_wsvb[1], w1_wsvb[2], w1_wsvb[3] };
        for (int n = 0; n < 64; ++n) {
            float sg = sigm(s1[n]);
            #pragma unroll
            for (int g = 0; g < 4; ++g) tg[g] = fmaf(sg, w1_wsv[n * 4 + g], tg[g]);
        }
        float v1[12];
        #pragma unroll
        for (int c = 0; c < 4; ++c) {
            float g = sigm(tg[c]);
            v1[c*3+0] = vp1[c*3+0] * g; v1[c*3+1] = vp1[c*3+1] * g; v1[c*3+2] = vp1[c*3+2] * g;
        }
        float vh2[12];
        #pragma unroll
        for (int h = 0; h < 4; ++h)
            #pragma unroll
            for (int d = 0; d < 3; ++d) {
                float t = 0.f;
                #pragma unroll
                for (int c = 0; c < 4; ++c) t = fmaf(v1[c*3+d], w2_wh[c*4+h], t);
                vh2[h*3+d] = t;
            }
        if (p == 0) {
            #pragma unroll
            for (int h = 0; h < 4; ++h) {
                float nn = vh2[h*3]*vh2[h*3] + vh2[h*3+1]*vh2[h*3+1] + vh2[h*3+2]*vh2[h*3+2];
                Ast[w][em][64 + h] = (__bf16)sqrtf(fmaxf(nn, 1e-8f));
            }
            #pragma unroll
            for (int c = 68; c < 80; ++c) Ast[w][em][c] = (__bf16)0.0f;
        } else {
            #pragma unroll
            for (int c = 80; c < 96; ++c) Ast[w][em][c] = (__bf16)0.0f;
        }
        #pragma unroll
        for (int c2 = 0; c2 < 4; ++c2)
            #pragma unroll
            for (int d = 0; d < 3; ++d) {
                float t = 0.f;
                #pragma unroll
                for (int h = 0; h < 4; ++h) t = fmaf(vh2[h*3+d], w2_wv[h*4+c2], t);
                vp2[c2*3+d] = t;
            }
    }
    __syncthreads();

    // ================= GVP layer 2: [16,96]x[96,64] ========================
    v8f bcc[4] = {};
    #pragma unroll
    for (int kt = 0; kt < 3; ++kt) {
        v16bf a = load_a(kt);
        #pragma unroll
        for (int nt = 0; nt < 4; ++nt) {
            v16bf b = *(const v16bf*)(w2f + (size_t)((kt * 4 + nt) * 32 + lane) * 16);
            bcc[nt] = __builtin_amdgcn_wmma_f32_16x16x32_bf16(
                false, a, false, b, (short)0, bcc[nt], false, false);
        }
    }
    #pragma unroll
    for (int nt = 0; nt < 4; ++nt)
        #pragma unroll
        for (int r = 0; r < 8; ++r) {
            int col = nt * 16 + mr, m = r + 8 * hi;
            float val = bcc[nt][r] + w2_wsb[col];
            ((float*)&Ast[w][m][128])[col] = val;               // raw s2
            Ast[w][m][col] = (__bf16)fmaxf(val, 0.f);           // relu(s2) -> A3
        }
    __syncthreads();

    // ---- per-edge: gate2, v2, vh3, vn3, vector partials for layer 3 ----
    float vp3[12];
    {
        const float* s2 = (const float*)&Ast[w][em][128];
        float tg[4] = { w2_wsvb[0], w2_wsvb[1], w2_wsvb[2], w2_wsvb[3] };
        for (int n = 0; n < 64; ++n) {
            float sg = sigm(s2[n]);
            #pragma unroll
            for (int g = 0; g < 4; ++g) tg[g] = fmaf(sg, w2_wsv[n * 4 + g], tg[g]);
        }
        float v2[12];
        #pragma unroll
        for (int c = 0; c < 4; ++c) {
            float g = sigm(tg[c]);
            v2[c*3+0] = vp2[c*3+0] * g; v2[c*3+1] = vp2[c*3+1] * g; v2[c*3+2] = vp2[c*3+2] * g;
        }
        float vh3[12];
        #pragma unroll
        for (int h = 0; h < 4; ++h)
            #pragma unroll
            for (int d = 0; d < 3; ++d) {
                float t = 0.f;
                #pragma unroll
                for (int c = 0; c < 4; ++c) t = fmaf(v2[c*3+d], w3_wh[c*4+h], t);
                vh3[h*3+d] = t;
            }
        if (p == 0) {
            #pragma unroll
            for (int h = 0; h < 4; ++h) {
                float nn = vh3[h*3]*vh3[h*3] + vh3[h*3+1]*vh3[h*3+1] + vh3[h*3+2]*vh3[h*3+2];
                Ast[w][em][64 + h] = (__bf16)sqrtf(fmaxf(nn, 1e-8f));
            }
            #pragma unroll
            for (int c = 68; c < 80; ++c) Ast[w][em][c] = (__bf16)0.0f;
        } else {
            #pragma unroll
            for (int c = 80; c < 96; ++c) Ast[w][em][c] = (__bf16)0.0f;
        }
        #pragma unroll
        for (int c2 = 0; c2 < 4; ++c2)
            #pragma unroll
            for (int d = 0; d < 3; ++d) {
                float t = 0.f;
                #pragma unroll
                for (int h = 0; h < 4; ++h) t = fmaf(vh3[h*3+d], w3_wv[h*4+c2], t);
                vp3[c2*3+d] = t;
            }
    }
    __syncthreads();

    // ================= GVP layer 3 (final): [16,96]x[96,64] ================
    v8f ccc[4] = {};
    #pragma unroll
    for (int kt = 0; kt < 3; ++kt) {
        v16bf a = load_a(kt);
        #pragma unroll
        for (int nt = 0; nt < 4; ++nt) {
            v16bf b = *(const v16bf*)(w3f + (size_t)((kt * 4 + nt) * 32 + lane) * 16);
            ccc[nt] = __builtin_amdgcn_wmma_f32_16x16x32_bf16(
                false, a, false, b, (short)0, ccc[nt], false, false);
        }
    }
    // raw s3 (no relu); also scatter scalar sums to node accumulators
    #pragma unroll
    for (int nt = 0; nt < 4; ++nt)
        #pragma unroll
        for (int r = 0; r < 8; ++r) {
            int col = nt * 16 + mr, m = r + 8 * hi;
            float val = ccc[nt][r] + w3_wsb[col];
            ((float*)&Ast[w][m][128])[col] = val;
            atomicAdd(&s_acc[(size_t)didx[w][m] * 64 + col], val);
        }
    __syncthreads();

    // ---- per-edge: final gate (no inner sigmoid), scatter vectors + count --
    {
        const float* s3 = (const float*)&Ast[w][em][128];
        float tg[4] = { w3_wsvb[0], w3_wsvb[1], w3_wsvb[2], w3_wsvb[3] };
        for (int n = 0; n < 64; ++n) {
            float sv = s3[n];
            #pragma unroll
            for (int g = 0; g < 4; ++g) tg[g] = fmaf(sv, w3_wsv[n * 4 + g], tg[g]);
        }
        #pragma unroll
        for (int cc = 0; cc < 2; ++cc) {
            int c = 2 * p + cc;
            float g = sigm(tg[c]);
            #pragma unroll
            for (int d = 0; d < 3; ++d)
                atomicAdd(&v_acc[(size_t)di * 12 + c * 3 + d], vp3[c * 3 + d] * g);
        }
        if (p == 0) atomicAdd(&cnt[di], 1.0f);
    }
}

// ---------------------------------------------------------------------------
// Finalize: mean, normalize vectors, cos/sin/exp scalars, apply mask.
__global__ void finalize_kernel(const float* __restrict__ s_acc,
                                const float* __restrict__ v_acc,
                                const float* __restrict__ cnt,
                                const unsigned char* __restrict__ mask,
                                float* __restrict__ out, int N_)
{
    int id = blockIdx.x * blockDim.x + threadIdx.x;
    int nsc = N_ * 64;
    if (id < nsc) {
        int n = id >> 6, j = id & 63;
        float denom = fmaxf(cnt[n], 1.0f);
        float s = s_acc[id] / denom;
        float r = (j < ANGS) ? cosf(s) : (j < 2 * ANGS) ? sinf(s) : expf(s);
        out[id] = mask[n] ? r : 0.0f;
    } else if (id < nsc + N_ * 4) {
        int t = id - nsc;
        int n = t >> 2, c = t & 3;
        float denom = fmaxf(cnt[n], 1.0f);
        float vx = v_acc[(size_t)n * 12 + c * 3 + 0] / denom;
        float vy = v_acc[(size_t)n * 12 + c * 3 + 1] / denom;
        float vz = v_acc[(size_t)n * 12 + c * 3 + 2] / denom;
        float mag = sqrtf(vx * vx + vy * vy + vz * vz);
        float sc = mask[n] ? (1.0f / (mag + 1e-8f)) : 0.0f;
        float* o = out + nsc + (size_t)n * 12 + c * 3;
        o[0] = vx * sc; o[1] = vy * sc; o[2] = vz * sc;
    }
}

// ---------------------------------------------------------------------------
extern "C" void kernel_launch(void* const* d_in, const int* in_sizes, int n_in,
                              void* d_out, int out_size, void* d_ws, size_t ws_size,
                              hipStream_t stream) {
    const float* node_s = (const float*)d_in[0];
    const float* node_v = (const float*)d_in[1];
    const int*   eidx   = (const int*)d_in[2];
    const float* edge_s = (const float*)d_in[3];
    const float* edge_v = (const float*)d_in[4];
    const unsigned char* mask = (const unsigned char*)d_in[5];
    const float* w1_wh = (const float*)d_in[6];
    const float* w1_ws = (const float*)d_in[7];
    const float* w1_wsb = (const float*)d_in[8];
    const float* w1_wv = (const float*)d_in[9];
    const float* w1_wsv = (const float*)d_in[10];
    const float* w1_wsvb = (const float*)d_in[11];
    const float* w2_wh = (const float*)d_in[12];
    const float* w2_ws = (const float*)d_in[13];
    const float* w2_wsb = (const float*)d_in[14];
    const float* w2_wv = (const float*)d_in[15];
    const float* w2_wsv = (const float*)d_in[16];
    const float* w2_wsvb = (const float*)d_in[17];
    const float* w3_wh = (const float*)d_in[18];
    const float* w3_ws = (const float*)d_in[19];
    const float* w3_wsb = (const float*)d_in[20];
    const float* w3_wv = (const float*)d_in[21];
    const float* w3_wsv = (const float*)d_in[22];
    const float* w3_wsvb = (const float*)d_in[23];

    const int N = in_sizes[0] / SI;    // 20000
    const int E = in_sizes[3] / SE;    // 320000

    // workspace layout
    uint8_t* ws = (uint8_t*)d_ws;
    float* s_acc = (float*)ws;                        // N*64 f32
    float* v_acc = s_acc + (size_t)N * 64;            // N*12 f32
    float* cntp  = v_acc + (size_t)N * 12;            // N f32
    size_t zbytes = (size_t)N * 77 * sizeof(float);
    size_t off = (zbytes + 255) & ~(size_t)255;       // 32B-aligned frag bases
    __bf16* w1f = (__bf16*)(ws + off);                // 11*4*32*16 bf16 = 45056 B
    __bf16* w2f = (__bf16*)(ws + off + 45056);        // 3*4*32*16 bf16 = 12288 B
    __bf16* w3f = (__bf16*)(ws + off + 45056 + 12288);

    // 1) zero accumulators
    int nz = N * 77;
    zero_f<<<(nz + 255) / 256, 256, 0, stream>>>(s_acc, nz);

    // 2) convert big weights to bf16 WMMA B fragments (zero-padded K)
    convert_w<<<(11 * 4 * 32 * 16 + 255) / 256, 256, 0, stream>>>(w1_ws, w1f, 11, 321);
    convert_w<<<(3 * 4 * 32 * 16 + 255) / 256, 256, 0, stream>>>(w2_ws, w2f, 3, 68);
    convert_w<<<(3 * 4 * 32 * 16 + 255) / 256, 256, 0, stream>>>(w3_ws, w3f, 3, 68);

    // 3) per-edge GVP pipeline (16 edges per wave, WAVES waves per block)
    int tiles = E / 16;
    dim3 grid(tiles / WAVES), block(32 * WAVES);
    gvp_edge_kernel<<<grid, block, 0, stream>>>(
        node_s, node_v, eidx, edge_s, edge_v,
        w1_wh, w1_wsb, w1_wv, w1_wsv, w1_wsvb,
        w2_wh, w2_wsb, w2_wv, w2_wsv, w2_wsvb,
        w3_wh, w3_wsb, w3_wv, w3_wsv, w3_wsvb,
        w1f, w2f, w3f, s_acc, v_acc, cntp, E);

    // 4) finalize per node
    int tot = N * 64 + N * 4;
    finalize_kernel<<<(tot + 255) / 256, 256, 0, stream>>>(s_acc, v_acc, cntp, mask,
                                                           (float*)d_out, N);
}
